// Pipeline_75935021793858
// MI455X (gfx1250) — matrix operand order repaired and checked by reading, not compiled, not run on hardware
//
#include <hip/hip_runtime.h>
#include <hip/hip_bf16.h>

typedef __bf16 bf16_t;
typedef __attribute__((ext_vector_type(16))) __bf16 v16bf;
typedef __attribute__((ext_vector_type(8)))  __bf16 v8bf;
typedef __attribute__((ext_vector_type(8)))  float  v8f;
typedef __attribute__((ext_vector_type(4)))  float  v4f;

#define NN 9216
#define CC 80
#define STEPS 40
#define ALPHA 0.9f

union bpack { v16bf v; v8bf h[2]; };

// ---------------------------------------------------------------------------
// P (fp32, row-major NxN) -> bf16, same layout. 8 elements per thread.
// P is read exactly once ever -> non-temporal loads.
// ---------------------------------------------------------------------------
__global__ __launch_bounds__(256)
void cvt_p_bf16(const float* __restrict__ P, bf16_t* __restrict__ Pb) {
  long i = ((long)blockIdx.x * blockDim.x + threadIdx.x) * 8;
  if (i >= (long)NN * NN) return;
  v4f f0 = __builtin_nontemporal_load((const v4f*)(P + i));
  v4f f1 = __builtin_nontemporal_load((const v4f*)(P + i + 4));
  v8bf o;
#pragma unroll
  for (int j = 0; j < 4; ++j) o[j] = (bf16_t)f0[j];
#pragma unroll
  for (int j = 0; j < 4; ++j) o[4 + j] = (bf16_t)f1[j];
  *(v8bf*)(Pb + i) = o;
}

// ---------------------------------------------------------------------------
// S0 (fp32, N x C row-major) -> bf16 transposed (C x N): Sb[c*N + n] = S0[n,c]
// ---------------------------------------------------------------------------
__global__ __launch_bounds__(256)
void init_s_bf16(const float* __restrict__ S0, bf16_t* __restrict__ Sb) {
  int i = blockIdx.x * blockDim.x + threadIdx.x;
  if (i >= CC * NN) return;
  int c = i / NN;
  int n = i - c * NN;
  Sb[i] = (bf16_t)S0[(long)n * CC + c];
}

// ---------------------------------------------------------------------------
// matched_score[c] = max over n of S0[n, c]
// ---------------------------------------------------------------------------
__global__ __launch_bounds__(256)
void col_max(const float* __restrict__ S0, float* __restrict__ out) {
  __shared__ float sm[256];
  const int c = blockIdx.x;
  float m = -3.4e38f;
  for (int n = threadIdx.x; n < NN; n += blockDim.x)
    m = fmaxf(m, S0[(long)n * CC + c]);
  sm[threadIdx.x] = m;
  __syncthreads();
  for (int s = 128; s > 0; s >>= 1) {
    if (threadIdx.x < s) sm[threadIdx.x] = fmaxf(sm[threadIdx.x], sm[threadIdx.x + s]);
    __syncthreads();
  }
  if (threadIdx.x == 0) out[c] = sm[0];
}

// ---------------------------------------------------------------------------
// One diffusion step: Snew = ALPHA * P @ S + (1-ALPHA) * S0, via bf16 WMMA.
//   Pb   : bf16 N x N row-major          (streamed, non-temporal)
//   Sin  : bf16 C x N (transposed S)     (L2-resident, regular temporal)
//   Sout : bf16 C x N (next transposed S)
//   Ffinal: if non-null, also store fp32 C x N (== trans_map output layout)
// Block = 128 threads = 4 waves; one 16-row M tile per block; 4-way K split
// across waves, combined in LDS.
// ---------------------------------------------------------------------------
__global__ __launch_bounds__(128)
void diffuse_step(const bf16_t* __restrict__ Pb,
                  const bf16_t* __restrict__ Sin,
                  const float*  __restrict__ S0,
                  bf16_t* __restrict__ Sout,
                  float*  __restrict__ Ffinal) {
  __shared__ float red[3 * 5 * 8 * 32];

  const int lane = threadIdx.x & 31;
  const int wave = threadIdx.x >> 5;
  const int m0   = blockIdx.x * 16;

  const int l15 = lane & 15;
  const int kg  = lane >> 4;     // 0 for lanes 0-15, 1 for lanes 16-31

  // A (16x32 bf16 tile of P): lane<16 holds row l15, K {0..7, 16..23};
  // lane>=16 holds row l15, K {8..15, 24..31}  -> base shifted by 8.
  const bf16_t* Abase = Pb + (long)(m0 + l15) * NN + kg * 8;
  // B (32x16 bf16 tile of S^T): lane<16 holds col l15, K {0..7, 16..23};
  // lane>=16 holds col l15, K {8..15, 24..31} -> base shifted by 8.
  const bf16_t* Bbase = Sin + (long)l15 * NN + kg * 8;

  v8f acc[5] = {};

  const int k0 = wave * (NN / 4);
  const int k1 = k0 + (NN / 4);
#pragma unroll 2
  for (int k = k0; k < k1; k += 32) {
    bpack a;
    // P is pure streaming traffic (170 MB/step > L2): non-temporal so it
    // does not evict the L2-resident S^T block.
    a.h[0] = __builtin_nontemporal_load((const v8bf*)(Abase + k));
    a.h[1] = __builtin_nontemporal_load((const v8bf*)(Abase + k + 16));
#pragma unroll
    for (int ct = 0; ct < 5; ++ct) {
      const bf16_t* bp = Bbase + (long)(ct * 16) * NN + k;
      bpack b;
      b.h[0] = *(const v8bf*)(bp);
      b.h[1] = *(const v8bf*)(bp + 16);
      acc[ct] = __builtin_amdgcn_wmma_f32_16x16x32_bf16(
          false, a.v, false, b.v, (short)0, acc[ct], false, false);
    }
  }

  // Combine the 4 K-partials through LDS (waves 1..3 dump, wave 0 sums).
  if (wave > 0) {
    float* dst = red + (wave - 1) * (5 * 8 * 32) + lane;
#pragma unroll
    for (int ct = 0; ct < 5; ++ct)
#pragma unroll
      for (int r = 0; r < 8; ++r)
        dst[(ct * 8 + r) * 32] = acc[ct][r];
  }
  __syncthreads();

  if (wave == 0) {
#pragma unroll
    for (int w = 0; w < 3; ++w) {
      const float* src = red + w * (5 * 8 * 32) + lane;
#pragma unroll
      for (int ct = 0; ct < 5; ++ct)
#pragma unroll
        for (int r = 0; r < 8; ++r)
          acc[ct][r] += src[(ct * 8 + r) * 32];
    }
    // D layout: lane<16 -> N=lane, M=r; lane>=16 -> N=lane-16, M=r+8.
#pragma unroll
    for (int ct = 0; ct < 5; ++ct) {
      const int c = ct * 16 + l15;
#pragma unroll
      for (int r = 0; r < 8; ++r) {
        const int m = m0 + r + kg * 8;
        float v = ALPHA * acc[ct][r] + (1.0f - ALPHA) * S0[(long)m * CC + c];
        Sout[(long)c * NN + m] = (bf16_t)v;
        if (Ffinal) Ffinal[(long)c * NN + m] = v;
      }
    }
  }
}

// ---------------------------------------------------------------------------
// Safety fallback (fp32, no big scratch needed): one thread per (n, c).
// ---------------------------------------------------------------------------
__global__ __launch_bounds__(256)
void naive_step(const float* __restrict__ P, const float* __restrict__ Sin,
                const float* __restrict__ S0, float* __restrict__ Sout,
                float* __restrict__ Ffinal) {
  int idx = blockIdx.x * blockDim.x + threadIdx.x;
  if (idx >= NN * CC) return;
  int n = idx / CC, c = idx - n * CC;
  const float* prow = P + (long)n * NN;
  float accum = 0.0f;
  for (int k = 0; k < NN; ++k) accum += prow[k] * Sin[(long)k * CC + c];
  float v = ALPHA * accum + (1.0f - ALPHA) * S0[idx];
  Sout[idx] = v;
  if (Ffinal) Ffinal[(long)c * NN + n] = v;
}

// ---------------------------------------------------------------------------
extern "C" void kernel_launch(void* const* d_in, const int* in_sizes, int n_in,
                              void* d_out, int out_size, void* d_ws, size_t ws_size,
                              hipStream_t stream) {
  (void)in_sizes; (void)n_in; (void)out_size;
  const float* P  = (const float*)d_in[0];
  const float* S0 = (const float*)d_in[1];
  float* trans   = (float*)d_out;                    // C*N fp32 trans_map
  float* matched = (float*)d_out + (long)CC * NN;    // C fp32 matched_score

  col_max<<<CC, 256, 0, stream>>>(S0, matched);

  const size_t needP = (size_t)NN * NN * sizeof(bf16_t);
  const size_t needS = ((size_t)CC * NN * sizeof(bf16_t) + 255) & ~(size_t)255;
  const size_t offA  = (needP + 255) & ~(size_t)255;
  const size_t offB  = offA + needS;

  if (ws_size >= offB + needS) {
    // bf16 WMMA path: stream P once per step at half the fp32 bytes.
    bf16_t* Pb = (bf16_t*)d_ws;
    bf16_t* Sa = (bf16_t*)((char*)d_ws + offA);
    bf16_t* Sb = (bf16_t*)((char*)d_ws + offB);

    const long pElems = (long)NN * NN;
    cvt_p_bf16<<<(unsigned)((pElems / 8 + 255) / 256), 256, 0, stream>>>(P, Pb);
    init_s_bf16<<<(CC * NN + 255) / 256, 256, 0, stream>>>(S0, Sa);

    for (int s = 0; s < STEPS; ++s) {
      bf16_t* in  = (s & 1) ? Sb : Sa;
      bf16_t* out = (s & 1) ? Sa : Sb;
      diffuse_step<<<NN / 16, 128, 0, stream>>>(
          Pb, in, S0, out, (s == STEPS - 1) ? trans : nullptr);
    }
  } else {
    // Fallback: fp32 double-buffered S in scratch.
    float* Sa = (float*)d_ws;
    float* Sb = Sa + (long)NN * CC;
    for (int s = 0; s < STEPS; ++s) {
      const float* in = (s == 0) ? S0 : ((s & 1) ? Sa : Sb);
      float* out = (s & 1) ? Sb : Sa;
      naive_step<<<(NN * CC + 255) / 256, 256, 0, stream>>>(
          P, in, S0, out, (s == STEPS - 1) ? trans : nullptr);
    }
  }
}
